// MultiHeadAttention_34900904247603
// MI455X (gfx1250) — compile-verified, matrix operand order checked by reading, not run on hardware
//
#include <hip/hip_runtime.h>
#include <cstdint>

// ---------------------------------------------------------------------------
// MI455X (gfx1250) causal MHA, bf16 WMMA pipeline.  B=4,T=2048,D=1024,H=16,HD=64
// Compute-bound (AI ~800 flop/byte) => v_wmma_f32_16x16x32_bf16 everywhere.
// GEMMs: async-to-LDS double-buffered B tile, raw split barriers (no vmem
// drain), register-pipelined A fragments, 2x unrolled ping-pong.
// Attention: flash-style, V pre-transposed.
// Workspace (88 MB):
//   [ 0MB) xb bf16 x 16MB | [16MB) wq/wk/wv/wo bf16 4x2MB | [24MB) Q 16MB
//   [40MB) K 16MB | [56MB) Vt [B,H,HD,T] 16MB | [72MB) AO [B*T,D] 16MB
// ---------------------------------------------------------------------------

#define DEV __device__ __forceinline__

typedef __attribute__((ext_vector_type(16))) __bf16 v16bf;
typedef __attribute__((ext_vector_type(8)))  float  v8f;
typedef __attribute__((ext_vector_type(4)))  int    v4i;

static constexpr int Bsz = 4, T = 2048, D = 1024, H = 16, HD = 64;
static constexpr int M = Bsz * T;           // 8192
static constexpr int BKP = 40;              // padded LDS row (32 + 8) -> 80B

union Frag16 { v16bf v; uint4 q[2]; };

// ---- CDNA5 async global->LDS path (ASYNCcnt), guarded --------------------
#if defined(__has_builtin)
#if __has_builtin(__builtin_amdgcn_global_load_async_to_lds_b128)
#define HAVE_ASYNC_LDS 1
#endif
#if __has_builtin(__builtin_amdgcn_s_barrier_signal) && \
    __has_builtin(__builtin_amdgcn_s_barrier_wait)
#define HAVE_SPLIT_BARRIER 1
#endif
#endif
#ifndef HAVE_ASYNC_LDS
#define HAVE_ASYNC_LDS 0
#endif
#ifndef HAVE_SPLIT_BARRIER
#define HAVE_SPLIT_BARRIER 0
#endif

DEV void copy16_to_lds(__bf16* ldst, const __bf16* gsrc) {
#if HAVE_ASYNC_LDS
  __builtin_amdgcn_global_load_async_to_lds_b128((v4i*)gsrc, (v4i*)ldst, 0, 0);
#else
  *(uint4*)ldst = *(const uint4*)gsrc;   // global_load_b128 + ds_store_b128
#endif
}
#if HAVE_ASYNC_LDS
#define ASYNC_WAIT_1() asm volatile("s_wait_asynccnt 0x1" ::: "memory")
#define ASYNC_WAIT_0() asm volatile("s_wait_asynccnt 0x0" ::: "memory")
#else
#define ASYNC_WAIT_1() do {} while (0)
#define ASYNC_WAIT_0() do {} while (0)
#endif

// Workgroup barrier without the full vmem drain of __syncthreads(): we manage
// ASYNCcnt/DScnt manually; asm fences keep the compiler from moving LDS ops
// across (the barrier intrinsics are IntrNoMem).
DEV void wg_barrier() {
#if HAVE_SPLIT_BARRIER
  asm volatile("" ::: "memory");
  __builtin_amdgcn_s_barrier_signal(-1);
  __builtin_amdgcn_s_barrier_wait(-1);
  asm volatile("" ::: "memory");
#else
  __syncthreads();
#endif
}
#if HAVE_SPLIT_BARRIER
#define DS_DRAIN() asm volatile("s_wait_dscnt 0x0" ::: "memory")
#else
#define DS_DRAIN() do {} while (0)
#endif

// 16x32 bf16 WMMA operand fragment (A-layout; B-operand passes the N-major
// matrix with ld=K). lanes 0-15 = rows; VGPR0-3: K{kg..kg+7},
// VGPR4-7: K{16+kg..16+kg+7}, kg = (lane>=16)*8.
DEV v16bf load_frag_g(const __bf16* p, int ld, int lane) {
  const int row = lane & 15;
  const int kg  = (lane >> 4) << 3;
  const __bf16* r0 = p + (size_t)row * ld + kg;
  Frag16 f;
  f.q[0] = *(const uint4*)(r0);
  f.q[1] = *(const uint4*)(r0 + 16);
  return f.v;
}

DEV v8f wmma_bf16(v16bf a, v16bf b, v8f c) {
  return __builtin_amdgcn_wmma_f32_16x16x32_bf16(
      false, a, false, b, (short)0, c, false, false);
}

// ---------------------------------------------------------------------------
__global__ __launch_bounds__(256) void cvt_f32_bf16(
    const float* __restrict__ src, __bf16* __restrict__ dst, int n) {
  int i = blockIdx.x * blockDim.x + threadIdx.x;
  if (i < n) dst[i] = (__bf16)src[i];
}

// ---------------------------------------------------------------------------
// NT GEMM: C[m,n] = sum_k A[m,k]*Bw[n,k].  A:[M,D], Bw:[D,D] bf16.
// WG 256 thr (8 waves). WG tile 256(M)x64(N); wave tile 32x64 (8 C-tiles).
// B tile (64x32) async-staged in LDS, double buffered, 16B/thread/k-step.
// mode 0: f32 [M,D] | mode 1: bf16 [B,H,T,HD] | mode 2: bf16 [B,H,HD,T]
// ---------------------------------------------------------------------------
__global__ __launch_bounds__(256) void gemm_bf16_nt(
    const __bf16* __restrict__ A, const __bf16* __restrict__ Bw,
    float* __restrict__ Cf, __bf16* __restrict__ Cb, int mode) {
  __shared__ __bf16 bTile[2][64][BKP];     // 2 x 64 x 40 bf16 = 10 KB

  const int tid  = threadIdx.x;
  const int lane = tid & 31;
  const int wave = tid >> 5;
  const int hi = lane >> 4, ln = lane & 15;
  const int m0 = blockIdx.x * 256 + wave * 32;
  const int n0 = blockIdx.y * 64;

  const int brow = tid >> 2;               // cooperative fill: row
  const int bchk = (tid & 3) * 8;          // 16B chunk within row

  const __bf16* aRow0 = A + (size_t)m0 * D;
  const __bf16* aRow1 = A + (size_t)(m0 + 16) * D;
  const __bf16* bSrc  = Bw + (size_t)(n0 + brow) * D + bchk;

  const v8f zero = {};
  v8f accA[4] = {zero, zero, zero, zero};
  v8f accB[4] = {zero, zero, zero, zero};

  const int NK = D / 32;   // 32 k-steps (even -> clean 2x unroll)

  // prologue: tile 0 -> buf0; A frags for k=0
  copy16_to_lds(&bTile[0][brow][bchk], bSrc);
  v16bf aA = load_frag_g(aRow0, D, lane);
  v16bf aB = load_frag_g(aRow1, D, lane);

  auto step = [&](int cur, int ks) {       // cur is a literal at each call
    const bool more = (ks + 1 < NK);
    const int kn = (ks + 1) * 32;
    v16bf nA, nB;
    if (more) {
      copy16_to_lds(&bTile[cur ^ 1][brow][bchk], bSrc + kn);
      __builtin_prefetch(aRow0 + kn + 32, 0, 1);
      nA = load_frag_g(aRow0 + kn, D, lane);
      nB = load_frag_g(aRow1 + kn, D, lane);
      ASYNC_WAIT_1();                      // tile ks landed, ks+1 in flight
    } else {
      ASYNC_WAIT_0();
    }
    wg_barrier();                          // tile ks visible to all waves

    const int kg = hi << 3;
#pragma unroll
    for (int nt = 0; nt < 4; ++nt) {
      Frag16 fb;
      fb.q[0] = *(const uint4*)&bTile[cur][nt * 16 + ln][kg];
      fb.q[1] = *(const uint4*)&bTile[cur][nt * 16 + ln][kg + 16];
      accA[nt] = wmma_bf16(aA, fb.v, accA[nt]);
      accB[nt] = wmma_bf16(aB, fb.v, accB[nt]);
    }
    DS_DRAIN();                            // own B-frag reads complete
    wg_barrier();                          // safe to overwrite buf[cur]
    aA = nA; aB = nB;
  };

  for (int ks = 0; ks < NK; ks += 2) {     // ping-pong, literal buffer index
    step(0, ks);
    step(1, ks + 1);
  }

#pragma unroll
  for (int half = 0; half < 2; ++half) {
#pragma unroll
    for (int nt = 0; nt < 4; ++nt) {
#pragma unroll
      for (int r = 0; r < 8; ++r) {
        const int m = m0 + half * 16 + r + hi * 8;
        const int n = n0 + nt * 16 + ln;
        const float v = half ? accB[nt][r] : accA[nt][r];
        if (mode == 0) {
          Cf[(size_t)m * D + n] = v;
        } else {
          const int b = m >> 11, t = m & (T - 1);
          const int h = n >> 6,  d = n & (HD - 1);
          if (mode == 1)
            Cb[((size_t)(b * H + h) * T + t) * HD + d] = (__bf16)v;
          else
            Cb[((size_t)(b * H + h) * HD + d) * T + t] = (__bf16)v;
        }
      }
    }
  }
}

// ---------------------------------------------------------------------------
// Flash attention, causal. grid=(B*H, T/128), block=256 (8 waves).
// Each wave owns 16 query rows; per-wave LDS staging for P (no block barriers
// in the causal loop -> divergent trip counts are safe). 8 WMMAs per 32-key
// block; V frag loads issued before the softmax VALU block.
// ---------------------------------------------------------------------------
__global__ __launch_bounds__(256) void flash_attn(
    const __bf16* __restrict__ Q, const __bf16* __restrict__ Kc,
    const __bf16* __restrict__ Vt, __bf16* __restrict__ AO) {
  __shared__ __bf16 pLds[8][16][BKP];      // 10 KB, one 16x32 tile per wave

  const int lane = threadIdx.x & 31;
  const int wave = threadIdx.x >> 5;
  const int hi = lane >> 4, ln = lane & 15;
  const int bh = blockIdx.x;
  const int bIdx = bh >> 4, h = bh & 15;
  const int q0 = blockIdx.y * 128 + wave * 16;

  const __bf16* Qbh = Q  + (size_t)bh * T * HD;
  const __bf16* Kbh = Kc + (size_t)bh * T * HD;
  const __bf16* Vbh = Vt + (size_t)bh * HD * T;

  const v16bf aq0 = load_frag_g(Qbh + (size_t)q0 * HD,      HD, lane);
  const v16bf aq1 = load_frag_g(Qbh + (size_t)q0 * HD + 32, HD, lane);

  v8f o0 = {}, o1 = {}, o2 = {}, o3 = {};
  float m_i[8], l_i[8];
#pragma unroll
  for (int r = 0; r < 8; ++r) { m_i[r] = -1e30f; l_i[r] = 0.f; }

  for (int kb = 0; kb <= q0 + 15; kb += 32) {
    // S = Q K^T for keys [kb, kb+32)
    v8f s0 = {}, s1 = {};
    s0 = wmma_bf16(aq0, load_frag_g(Kbh + (size_t)kb * HD,             HD, lane), s0);
    s0 = wmma_bf16(aq1, load_frag_g(Kbh + (size_t)kb * HD + 32,        HD, lane), s0);
    s1 = wmma_bf16(aq0, load_frag_g(Kbh + (size_t)(kb + 16) * HD,      HD, lane), s1);
    s1 = wmma_bf16(aq1, load_frag_g(Kbh + (size_t)(kb + 16) * HD + 32, HD, lane), s1);

    // issue V fragments now; softmax VALU below hides their latency
    const v16bf bv0 = load_frag_g(Vbh + (size_t)0  * T + kb, T, lane);
    const v16bf bv1 = load_frag_g(Vbh + (size_t)16 * T + kb, T, lane);
    const v16bf bv2 = load_frag_g(Vbh + (size_t)32 * T + kb, T, lane);
    const v16bf bv3 = load_frag_g(Vbh + (size_t)48 * T + kb, T, lane);

    // online softmax; row stats replicated across each 16-lane half
#pragma unroll
    for (int r = 0; r < 8; ++r) {
      const int qrow = q0 + r + hi * 8;
      float x0 = s0[r] * 0.125f; if (kb + ln > qrow)      x0 = -1e30f;
      float x1 = s1[r] * 0.125f; if (kb + 16 + ln > qrow) x1 = -1e30f;
      float mx = fmaxf(x0, x1);
      mx = fmaxf(mx, __shfl_xor(mx, 1));
      mx = fmaxf(mx, __shfl_xor(mx, 2));
      mx = fmaxf(mx, __shfl_xor(mx, 4));
      mx = fmaxf(mx, __shfl_xor(mx, 8));
      const float mn = fmaxf(m_i[r], mx);
      const float p0 = __expf(x0 - mn);
      const float p1 = __expf(x1 - mn);
      float rs = p0 + p1;
      rs += __shfl_xor(rs, 1); rs += __shfl_xor(rs, 2);
      rs += __shfl_xor(rs, 4); rs += __shfl_xor(rs, 8);
      const float alpha = __expf(m_i[r] - mn);
      m_i[r] = mn;
      l_i[r] = l_i[r] * alpha + rs;
      o0[r] *= alpha; o1[r] *= alpha; o2[r] *= alpha; o3[r] *= alpha;
      pLds[wave][r + hi * 8][ln]      = (__bf16)p0;   // C layout -> LDS
      pLds[wave][r + hi * 8][16 + ln] = (__bf16)p1;
    }
    asm volatile("" ::: "memory");   // keep DS load after DS stores (same wave)

    // re-read P in A-fragment layout (same-wave DS ops are in order)
    Frag16 fp;
    fp.q[0] = *(const uint4*)&pLds[wave][ln][hi * 8];
    fp.q[1] = *(const uint4*)&pLds[wave][ln][hi * 8 + 16];
    const v16bf ap = fp.v;

    o0 = wmma_bf16(ap, bv0, o0);
    o1 = wmma_bf16(ap, bv1, o1);
    o2 = wmma_bf16(ap, bv2, o2);
    o3 = wmma_bf16(ap, bv3, o3);
  }

#pragma unroll
  for (int r = 0; r < 8; ++r) {
    const float inv = 1.0f / l_i[r];
    const int t = q0 + r + hi * 8;
    const size_t row = ((size_t)bIdx * T + t) * D + h * HD;
    AO[row +  0 + ln] = (__bf16)(o0[r] * inv);
    AO[row + 16 + ln] = (__bf16)(o1[r] * inv);
    AO[row + 32 + ln] = (__bf16)(o2[r] * inv);
    AO[row + 48 + ln] = (__bf16)(o3[r] * inv);
  }
}

// ---------------------------------------------------------------------------
extern "C" void kernel_launch(void* const* d_in, const int* in_sizes, int n_in,
                              void* d_out, int out_size, void* d_ws, size_t ws_size,
                              hipStream_t stream) {
  const float* x  = (const float*)d_in[0];
  const float* Wq = (const float*)d_in[1];
  const float* Wk = (const float*)d_in[2];
  const float* Wv = (const float*)d_in[3];
  const float* Wo = (const float*)d_in[4];
  float* out = (float*)d_out;

  char* ws = (char*)d_ws;
  __bf16* xb  = (__bf16*)(ws);
  __bf16* wqb = (__bf16*)(ws + (16ull << 20));
  __bf16* wkb = (__bf16*)(ws + (18ull << 20));
  __bf16* wvb = (__bf16*)(ws + (20ull << 20));
  __bf16* wob = (__bf16*)(ws + (22ull << 20));
  __bf16* Qb  = (__bf16*)(ws + (24ull << 20));
  __bf16* Kb  = (__bf16*)(ws + (40ull << 20));
  __bf16* Vtb = (__bf16*)(ws + (56ull << 20));
  __bf16* AOb = (__bf16*)(ws + (72ull << 20));

  const int nX = M * D;
  const int nW = D * D;
  cvt_f32_bf16<<<(nX + 255) / 256, 256, 0, stream>>>(x,  xb,  nX);
  cvt_f32_bf16<<<(nW + 255) / 256, 256, 0, stream>>>(Wq, wqb, nW);
  cvt_f32_bf16<<<(nW + 255) / 256, 256, 0, stream>>>(Wk, wkb, nW);
  cvt_f32_bf16<<<(nW + 255) / 256, 256, 0, stream>>>(Wv, wvb, nW);
  cvt_f32_bf16<<<(nW + 255) / 256, 256, 0, stream>>>(Wo, wob, nW);

  dim3 ggrid(M / 256, D / 64);   // (32, 16)
  gemm_bf16_nt<<<ggrid, 256, 0, stream>>>(xb, wqb, nullptr, Qb,  1);
  gemm_bf16_nt<<<ggrid, 256, 0, stream>>>(xb, wkb, nullptr, Kb,  1);
  gemm_bf16_nt<<<ggrid, 256, 0, stream>>>(xb, wvb, nullptr, Vtb, 2);

  dim3 fgrid(Bsz * H, T / 128);  // (64, 16)
  flash_attn<<<fgrid, 256, 0, stream>>>(Qb, Kb, Vtb, AOb);

  gemm_bf16_nt<<<ggrid, 256, 0, stream>>>(AOb, wob, out, nullptr, 0);
}
